// MoEModel_42116449305004
// MI455X (gfx1250) — matrix operand order repaired and checked by reading, not compiled, not run on hardware
//
#include <hip/hip_runtime.h>
#include <hip/hip_bf16.h>

// ---------------- problem constants ----------------
#define BATCH    8192
#define IN_DIM   1024
#define HID      256
#define ODIM     256
#define NEXP     64
#define TOPK     24
#define M_TILE   64
#define BK       32      // WMMA K-depth for f16

typedef __attribute__((ext_vector_type(16))) _Float16 v16h;
typedef __attribute__((ext_vector_type(8)))  _Float16 v8h;
typedef __attribute__((ext_vector_type(8)))  float    v8f;

__device__ __forceinline__ v16h cat8(v8h lo, v8h hi) {
    return __builtin_shufflevector(lo, hi, 0,1,2,3,4,5,6,7,8,9,10,11,12,13,14,15);
}

__device__ __forceinline__ v8f wmma_f16(v16h a, v16h b, v8f c) {
    // emits v_wmma_f32_16x16x32_f16
    return __builtin_amdgcn_wmma_f32_16x16x32_f16(
        /*neg_a=*/false, a, /*neg_b=*/false, b,
        /*c_mod=*/(short)0, c, /*reuse_a=*/false, /*reuse_b=*/false);
}

// async 16-byte global -> LDS copy (CDNA5 GLOBAL_LOAD_ASYNC_TO_LDS_B128,
// tracked by ASYNCcnt). ldsoff = wave-relative LDS byte address.
__device__ __forceinline__ void async_g2l_b128(unsigned ldsoff, const void* gaddr) {
    asm volatile("global_load_async_to_lds_b128 %0, %1, off"
                 :: "v"(ldsoff), "v"(gaddr) : "memory");
}
// wait until ASYNCcnt <= N (async loads complete in order, so N>0 waits for
// the oldest outstanding chunk while newer ones stay in flight)
template <int N>
__device__ __forceinline__ void wait_async_le() {
    asm volatile("s_wait_asynccnt %0" :: "n"(N) : "memory");
}
__device__ __forceinline__ unsigned lds_addr_of(const void* p) {
    return (unsigned)(uintptr_t)p;   // flat->LDS: low 32 bits are the LDS offset
}

// ---------------- kernel 1a: linear f32 -> f16 convert (activations) ------
__global__ void f32_to_f16_kernel(const float* __restrict__ src,
                                  _Float16* __restrict__ dst, int n) {
    int i = blockIdx.x * blockDim.x + threadIdx.x;
    int stride = gridDim.x * blockDim.x;
    for (; i < n; i += stride) dst[i] = (_Float16)src[i];
}

// ---------------- kernel 1b: convert + transpose (weights) ----------------
// src: [e][R][C] f32 (row-major), dst: [e][C][R] f16 (transposed)
// grid (C/32, R/32, NEXP), block 256 = 32x8
__global__ void convert_transpose_kernel(const float* __restrict__ src,
                                         _Float16* __restrict__ dst,
                                         int R, int C) {
    __shared__ _Float16 tile[32][33];   // +1 pad to avoid bank conflicts
    int e  = blockIdx.z;
    int c0 = blockIdx.x * 32;
    int r0 = blockIdx.y * 32;
    int tx = threadIdx.x & 31;
    int ty = threadIdx.x >> 5;          // 0..7
    const float* s = src + ((size_t)e * R + r0) * C + c0;
    #pragma unroll
    for (int i = ty; i < 32; i += 8)
        tile[i][tx] = (_Float16)s[(size_t)i * C + tx];      // coalesced read
    __syncthreads();
    _Float16* d = dst + ((size_t)e * C + c0) * R + r0;
    #pragma unroll
    for (int i = ty; i < 32; i += 8)
        d[(size_t)i * R + tx] = tile[tx][i];                // coalesced write
}

// ---------------- kernel 2: zero output + expert counters ----------------
__global__ void zero_kernel(float* __restrict__ out, int n,
                            int* __restrict__ counts, int ne) {
    int i = blockIdx.x * blockDim.x + threadIdx.x;
    int stride = gridDim.x * blockDim.x;
    for (int j = i; j < n; j += stride) out[j] = 0.0f;
    if (i < ne) counts[i] = 0;
}

// ---------------- kernel 3: gate logits + top-k + softmax ----------------
// one 64-thread block per token
__global__ void gate_kernel(const float* __restrict__ x,
                            const float* __restrict__ gw,   // [IN_DIM, NEXP]
                            const float* __restrict__ gb,   // [NEXP]
                            int* __restrict__ tidx,         // [BATCH, TOPK]
                            float* __restrict__ tw) {       // [BATCH, TOPK]
    __shared__ float xr[IN_DIM];
    __shared__ float logits[NEXP];
    int b = blockIdx.x;
    int t = threadIdx.x;   // 0..63, one expert per thread
    for (int i = t; i < IN_DIM; i += 64)
        xr[i] = x[(size_t)b * IN_DIM + i];
    __syncthreads();
    float s = gb[t];
    for (int i = 0; i < IN_DIM; ++i)
        s = fmaf(xr[i], gw[i * NEXP + t], s);
    logits[t] = s;
    __syncthreads();
    if (t == 0) {
        unsigned long long used = 0ull;
        int   sel[TOPK];
        float vals[TOPK];
        for (int j = 0; j < TOPK; ++j) {
            int bi = 0; float bv = -INFINITY;
            for (int i2 = 0; i2 < NEXP; ++i2) {
                if (!((used >> i2) & 1ull) && logits[i2] > bv) { bv = logits[i2]; bi = i2; }
            }
            used |= (1ull << bi);
            sel[j] = bi; vals[j] = bv;
        }
        float m = vals[0];            // descending order -> first is max
        float sum = 0.0f;
        for (int j = 0; j < TOPK; ++j) { vals[j] = __expf(vals[j] - m); sum += vals[j]; }
        float inv = 1.0f / sum;
        for (int j = 0; j < TOPK; ++j) {
            tidx[(size_t)b * TOPK + j] = sel[j];
            tw  [(size_t)b * TOPK + j] = vals[j] * inv;
        }
    }
}

// ---------------- kernel 4: build per-expert token lists ----------------
__global__ void scatter_kernel(const int* __restrict__ tidx,
                               const float* __restrict__ tw,
                               int* __restrict__ counts,
                               int* __restrict__ lists,     // [NEXP, BATCH]
                               float* __restrict__ listw) { // [NEXP, BATCH]
    int i = blockIdx.x * blockDim.x + threadIdx.x;
    if (i >= BATCH * TOPK) return;
    int b = i / TOPK;
    int e = tidx[i];
    int pos = atomicAdd(&counts[e], 1);
    lists[e * BATCH + pos] = b;
    listw[e * BATCH + pos] = tw[i];
}

// ---------------- kernel 5: fused expert MLP via WMMA ----------------
// grid (BATCH/M_TILE, NEXP), 256 threads = 8 waves (wave32)
// wave w: wm = w>>1 (row group of 16), wn = w&1 (col half of 128)
// weights arrive PRE-TRANSPOSED: w1t [e][HID][IN_DIM], w2t [e][ODIM][HID]
// staging: double-buffered async global->LDS (ASYNCcnt pipelining).
__global__ void __launch_bounds__(256)
expert_kernel(const _Float16* __restrict__ xh,   // [BATCH, IN_DIM]
              const _Float16* __restrict__ w1t,  // [NEXP, HID, IN_DIM]
              const float* __restrict__ b1,      // [NEXP, HID]
              const _Float16* __restrict__ w2t,  // [NEXP, ODIM, HID]
              const float* __restrict__ b2,      // [NEXP, ODIM]
              const int* __restrict__ counts,
              const int* __restrict__ lists,
              const float* __restrict__ listw,
              float* __restrict__ out) {         // [BATCH, ODIM]
    int e   = blockIdx.y;
    int cnt = counts[e];
    int m0  = blockIdx.x * M_TILE;
    if (m0 >= cnt) return;

    __shared__ __align__(16) _Float16 Xs[2][M_TILE][BK];   // 2 x 4 KB
    __shared__ __align__(16) _Float16 Wt[2][HID][BK];      // 2 x 16 KB, [col][k]
    __shared__ __align__(16) _Float16 H [M_TILE][HID];     // 32 KB hidden (f16)
    __shared__ int   toks[M_TILE];
    __shared__ float wgts[M_TILE];

    int tid = threadIdx.x;
    if (tid < M_TILE) {
        int r = m0 + tid;
        toks[tid] = (r < cnt) ? lists[e * BATCH + r] : lists[e * BATCH + m0];
        wgts[tid] = (r < cnt) ? listw[e * BATCH + r] : 0.0f;
    }
    __syncthreads();

    int wave = tid >> 5;
    int lane = tid & 31;
    int wm   = wave >> 1;        // 0..3
    int wn   = wave & 1;         // 0..1
    int lrow = lane & 15;
    int lhi  = lane >> 4;        // 0/1

    // ---- per-thread staging descriptors ----
    // X: thread -> (row tid>>2, 8-half segment tid&3)
    int xr_  = tid >> 2, xseg = tid & 3;
    const _Float16* xsrc = xh + (size_t)toks[xr_] * IN_DIM + xseg * 8;
    unsigned xs_lds[2] = { lds_addr_of(&Xs[0][xr_][xseg * 8]),
                           lds_addr_of(&Xs[1][xr_][xseg * 8]) };
    // W: thread stages the whole 32-half column `tid` (4 x b128)
    unsigned wt_lds[2] = { lds_addr_of(&Wt[0][tid][0]),
                           lds_addr_of(&Wt[1][tid][0]) };
    const _Float16* w1e = w1t + (size_t)e * HID * IN_DIM;   // [HID][IN_DIM]
    const _Float16* w1c = w1e + (size_t)tid * IN_DIM;       // column tid
    const _Float16* w2e = w2t + (size_t)e * ODIM * HID;     // [ODIM][HID]
    const _Float16* w2c = w2e + (size_t)tid * HID;

    v8f acc[8] = {};

    // ---------------- GEMM1: [64 x 1024] @ [1024 x 256] ----------------
    const int NC1 = IN_DIM / BK;    // 32 chunks, 5 asyncs/thread/chunk
    // prologue: chunk 0 -> buffer 0
    async_g2l_b128(xs_lds[0], xsrc);
    #pragma unroll
    for (int j = 0; j < 4; ++j)
        async_g2l_b128(wt_lds[0] + j * 16, w1c + j * 8);

    int pb = 0;
    for (int i = 0; i < NC1; ++i) {
        int k0 = i * BK;
        if (i + 1 < NC1) {
            // issue chunk i+1 into the other buffer, then wait for chunk i
            async_g2l_b128(xs_lds[pb ^ 1], xsrc + k0 + BK);
            #pragma unroll
            for (int j = 0; j < 4; ++j)
                async_g2l_b128(wt_lds[pb ^ 1] + j * 16, w1c + k0 + BK + j * 8);
            wait_async_le<5>();
        } else {
            wait_async_le<0>();
        }
        __syncthreads();

        // A fragment: lanes0-15 K={0..7,16..23}, lanes16-31 K={8..15,24..31}
        v8h alo = *(const v8h*)&Xs[pb][wm * 16 + lrow][lhi * 8];
        v8h ahi = *(const v8h*)&Xs[pb][wm * 16 + lrow][16 + lhi * 8];
        v16h afrag = cat8(alo, ahi);
        #pragma unroll
        for (int c = 0; c < 8; ++c) {
            int col = wn * 128 + c * 16 + lrow;
            // B fragment: lanes0-15 K={0..7,16..23}, lanes16-31 K={8..15,24..31}
            v8h blo = *(const v8h*)&Wt[pb][col][lhi * 8];
            v8h bhi = *(const v8h*)&Wt[pb][col][16 + lhi * 8];
            acc[c] = wmma_f16(afrag, cat8(blo, bhi), acc[c]);
        }
        __syncthreads();
        pb ^= 1;
    }

    // GEMM2 prologue: stream first W2 chunk while we run epilogue-1
    // (buffer 0 readers finished two barriers ago -> safe)
    #pragma unroll
    for (int j = 0; j < 4; ++j)
        async_g2l_b128(wt_lds[0] + j * 16, w2c + j * 8);

    // epilogue 1: bias + ReLU -> H (f16)
    #pragma unroll
    for (int c = 0; c < 8; ++c) {
        int col = wn * 128 + c * 16 + lrow;
        float bias = b1[e * HID + col];
        #pragma unroll
        for (int v = 0; v < 8; ++v) {
            int row = wm * 16 + lhi * 8 + v;
            float h = acc[c][v] + bias;
            H[row][col] = (_Float16)(h > 0.0f ? h : 0.0f);
        }
    }
    __syncthreads();

    // ---------------- GEMM2: [64 x 256] @ [256 x 256] ----------------
    v8f acc2[8] = {};
    const int NC2 = HID / BK;       // 8 chunks, 4 asyncs/thread/chunk
    pb = 0;
    for (int i = 0; i < NC2; ++i) {
        int k0 = i * BK;
        if (i + 1 < NC2) {
            #pragma unroll
            for (int j = 0; j < 4; ++j)
                async_g2l_b128(wt_lds[pb ^ 1] + j * 16, w2c + k0 + BK + j * 8);
            wait_async_le<4>();
        } else {
            wait_async_le<0>();
        }
        __syncthreads();

        v8h alo = *(const v8h*)&H[wm * 16 + lrow][k0 + lhi * 8];
        v8h ahi = *(const v8h*)&H[wm * 16 + lrow][k0 + 16 + lhi * 8];
        v16h afrag = cat8(alo, ahi);
        #pragma unroll
        for (int c = 0; c < 8; ++c) {
            int col = wn * 128 + c * 16 + lrow;
            v8h blo = *(const v8h*)&Wt[pb][col][lhi * 8];
            v8h bhi = *(const v8h*)&Wt[pb][col][16 + lhi * 8];
            acc2[c] = wmma_f16(afrag, cat8(blo, bhi), acc2[c]);
        }
        __syncthreads();
        pb ^= 1;
    }

    // epilogue 2: bias, routing weight, scatter-accumulate to out
    #pragma unroll
    for (int c = 0; c < 8; ++c) {
        int col = wn * 128 + c * 16 + lrow;
        float bias = b2[e * ODIM + col];
        #pragma unroll
        for (int v = 0; v < 8; ++v) {
            int row = wm * 16 + lhi * 8 + v;
            if (m0 + row < cnt) {
                float val = (acc2[c][v] + bias) * wgts[row];
                unsafeAtomicAdd(&out[(size_t)toks[row] * ODIM + col], val);
            }
        }
    }
}

// ---------------- host launcher ----------------
extern "C" void kernel_launch(void* const* d_in, const int* in_sizes, int n_in,
                              void* d_out, int out_size, void* d_ws, size_t ws_size,
                              hipStream_t stream) {
    const float* x  = (const float*)d_in[0];
    const float* gw = (const float*)d_in[1];
    const float* gb = (const float*)d_in[2];
    const float* w1 = (const float*)d_in[3];
    const float* b1 = (const float*)d_in[4];
    const float* w2 = (const float*)d_in[5];
    const float* b2 = (const float*)d_in[6];
    // d_in[7] = k (compile-time TOPK=24)
    float* out = (float*)d_out;

    char* ws = (char*)d_ws;
    _Float16* xh  = (_Float16*)ws;  ws += (size_t)BATCH * IN_DIM * 2;
    _Float16* w1t = (_Float16*)ws;  ws += (size_t)NEXP * IN_DIM * HID * 2;   // [e][HID][IN_DIM]
    _Float16* w2t = (_Float16*)ws;  ws += (size_t)NEXP * HID * ODIM * 2;     // [e][ODIM][HID]
    int*   tidx   = (int*)ws;       ws += (size_t)BATCH * TOPK * 4;
    float* tw     = (float*)ws;     ws += (size_t)BATCH * TOPK * 4;
    int*   counts = (int*)ws;       ws += 256;
    int*   lists  = (int*)ws;       ws += (size_t)NEXP * BATCH * 4;
    float* listw  = (float*)ws;     ws += (size_t)NEXP * BATCH * 4;

    f32_to_f16_kernel<<<2048, 256, 0, stream>>>(x, xh, BATCH * IN_DIM);
    // w1 [e][1024][256] -> w1t [e][256][1024]
    convert_transpose_kernel<<<dim3(HID / 32, IN_DIM / 32, NEXP), 256, 0, stream>>>(
        w1, w1t, IN_DIM, HID);
    // w2 [e][256][256] -> w2t [e][256][256]
    convert_transpose_kernel<<<dim3(ODIM / 32, HID / 32, NEXP), 256, 0, stream>>>(
        w2, w2t, HID, ODIM);
    zero_kernel<<<2048, 256, 0, stream>>>(out, BATCH * ODIM, counts, NEXP);
    gate_kernel<<<BATCH, 64, 0, stream>>>(x, gw, gb, tidx, tw);
    scatter_kernel<<<(BATCH * TOPK + 255) / 256, 256, 0, stream>>>(tidx, tw, counts, lists, listw);
    expert_kernel<<<dim3(BATCH / M_TILE, NEXP), 256, 0, stream>>>(
        xh, w1t, b1, w2t, b2, counts, lists, listw, out);
}
